// MultiHeadAttention_71940702208560
// MI455X (gfx1250) — matrix operand order repaired and checked by reading, not compiled, not run on hardware
//
#include <hip/hip_runtime.h>

// MI455X / gfx1250: wave32, WMMA 16x16x32 bf16 (f32 accumulate),
// async global->LDS staging (ASYNCcnt).
typedef __attribute__((ext_vector_type(16))) __bf16       v16bf;
typedef __attribute__((ext_vector_type(8)))  float        v8f;
typedef __attribute__((ext_vector_type(8)))  unsigned int v8ui;
typedef int v4i __attribute__((vector_size(16)));

#define B_   2
#define S_   2048
#define D_   1024
#define H_   16
#define HD_  64
#define M_   (B_ * S_)      // 4096
#define N3_  (3 * D_)       // 3072

// ---- async global->LDS 16B copy (per-lane addresses) ----
__device__ __forceinline__ void async_g2l_16B(void* lds, const void* g) {
#if __has_builtin(__builtin_amdgcn_global_load_async_to_lds_b128)
  __builtin_amdgcn_global_load_async_to_lds_b128(
      (__attribute__((address_space(1))) v4i*)(g),
      (__attribute__((address_space(3))) v4i*)(lds), 0, 0);
#else
  unsigned loff =
      (unsigned)(size_t)(__attribute__((address_space(3))) void*)lds;
  asm volatile("global_load_async_to_lds_b128 %0, %1, off"
               :: "v"(loff), "v"((unsigned long long)(size_t)g)
               : "memory");
#endif
}

__device__ __forceinline__ void wait_async0() {
#if __has_builtin(__builtin_amdgcn_s_wait_asynccnt)
  __builtin_amdgcn_s_wait_asynccnt(0);
#else
  asm volatile("s_wait_asynccnt 0x0" ::: "memory");
#endif
}

// A fragment (16x32 bf16): lane-half h, row m: elements i=0..7  -> k = h*8+i,
// i=8..15 -> k = 16+h*8+(i-8).  Both halves are contiguous 16B runs in a
// row-major [m][32] tile -> 2x ds_load_b128.
__device__ __forceinline__ v16bf load_a_frag(const __bf16* rowk, int h) {
  uint4 lo = *(const uint4*)(rowk + h * 8);
  uint4 hi = *(const uint4*)(rowk + 16 + h * 8);
  v8ui r;
  r[0] = lo.x; r[1] = lo.y; r[2] = lo.z; r[3] = lo.w;
  r[4] = hi.x; r[5] = hi.y; r[6] = hi.z; r[7] = hi.w;
  return __builtin_bit_cast(v16bf, r);
}

// B fragment (32x16 bf16): lane n, half h: elements i=0..7 -> k = h*8+i,
// i=8..15 -> k = 16+h*8+(i-8) (p points at row + h*16).
__device__ __forceinline__ v16bf load_b_frag(const __bf16* p) {
  const int hb = (threadIdx.x >> 4) & 1;
  const __bf16* rowk = p - hb * 8;
  uint4 lo = *(const uint4*)(rowk);
  uint4 hi = *(const uint4*)(rowk + 16);
  v8ui r;
  r[0] = lo.x; r[1] = lo.y; r[2] = lo.z; r[3] = lo.w;
  r[4] = hi.x; r[5] = hi.y; r[6] = hi.z; r[7] = hi.w;
  return __builtin_bit_cast(v16bf, r);
}

__global__ void cvt_f32_bf16(const float* __restrict__ src,
                             __bf16* __restrict__ dst, int n) {
  int i = blockIdx.x * blockDim.x + threadIdx.x;
  int stride = gridDim.x * blockDim.x;
  for (; i < n; i += stride) dst[i] = (__bf16)src[i];
}

// dst[n][k] = (bf16) src[k][n].  32x32 tiles through LDS; coalesced both ways.
__global__ __launch_bounds__(256) void transpose_f32_bf16(
    const float* __restrict__ src, __bf16* __restrict__ dst, int K, int N) {
  __shared__ __align__(16) __bf16 lt[32 * 40];   // row stride 40 (80B, 16B-aligned)
  const int n0 = blockIdx.x * 32, k0 = blockIdx.y * 32;
  const int t = threadIdx.x;
  const int r = t >> 3, c4 = (t & 7) << 2;       // 32 rows x 8 float4 chunks
  float4 f = *(const float4*)(src + (size_t)(k0 + r) * N + n0 + c4);
  lt[(c4 + 0) * 40 + r] = (__bf16)f.x;
  lt[(c4 + 1) * 40 + r] = (__bf16)f.y;
  lt[(c4 + 2) * 40 + r] = (__bf16)f.z;
  lt[(c4 + 3) * 40 + r] = (__bf16)f.w;
  __syncthreads();
  if (t < 128) {
    int rr = t >> 2, cc8 = (t & 3) << 3;
    *(uint4*)(dst + (size_t)(n0 + rr) * K + k0 + cc8) =
        *(const uint4*)&lt[rr * 40 + cc8];
  }
}

// C = A(bf16 MxK row-major) @ B where BT is bf16 [N][K] (pre-transposed).
// 8 waves, each 32x32 (2x2 frags): per k-step 3 async-to-LDS + 8 ds_load_b128
// feed 4 WMMA.  mode 0: +bias, scatter q/k [B,H,S,64] and vT [B,H,64,S] bf16.
// mode 1: +bias, f32 to outF[m*N+n].
__global__ __launch_bounds__(256) void gemm_bf16_wmma(
    const __bf16* __restrict__ A, const __bf16* __restrict__ BT,
    const float* __restrict__ bias, int Ndim, int Kdim, int mode,
    float* __restrict__ outF, __bf16* __restrict__ q,
    __bf16* __restrict__ k, __bf16* __restrict__ vT) {
  __shared__ __align__(16) __bf16 lA[64 * 32];    //  4 KB, row-major [m][k]
  __shared__ __align__(16) __bf16 lB[128 * 32];   //  8 KB, n-major  [n][k]

  const int tid  = threadIdx.x;
  const int lane = tid & 31;
  const int h    = lane >> 4;
  const int nl   = lane & 15;
  const int wid  = tid >> 5;       // 8 waves: 2 (m) x 4 (n)
  const int wm   = wid & 1;
  const int wn   = wid >> 1;
  const int n0   = blockIdx.x * 128;
  const int m0   = blockIdx.y * 64;

  // per-thread 16B chunk coordinates (each wave issues 1 async for lA, 2 for lB)
  const int iiA = tid;                 // 256 chunks: 64 rows x 4
  const int rA  = iiA >> 2, cA = (iiA & 3) << 3;
  const int iiB0 = tid, iiB1 = tid + 256;          // 512 chunks: 128 rows x 4
  const int rB0 = iiB0 >> 2, cB0 = (iiB0 & 3) << 3;
  const int rB1 = iiB1 >> 2, cB1 = (iiB1 & 3) << 3;

  v8f c[2][2] = {};

  for (int k0 = 0; k0 < Kdim; k0 += 32) {
    async_g2l_16B((char*)lA + iiA * 16,
                  A + (size_t)(m0 + rA) * Kdim + k0 + cA);
    async_g2l_16B((char*)lB + iiB0 * 16,
                  BT + (size_t)(n0 + rB0) * Kdim + k0 + cB0);
    async_g2l_16B((char*)lB + iiB1 * 16,
                  BT + (size_t)(n0 + rB1) * Kdim + k0 + cB1);
    if (k0 + 32 < Kdim) {
      __builtin_prefetch(A + (size_t)(m0 + rA) * Kdim + k0 + 32 + cA, 0, 1);
      __builtin_prefetch(BT + (size_t)(n0 + rB0) * Kdim + k0 + 32 + cB0, 0, 1);
    }
    wait_async0();
    __syncthreads();

    v16bf a[2], b[2];
#pragma unroll
    for (int mt = 0; mt < 2; ++mt)
      a[mt] = load_a_frag(&lA[(wm * 32 + mt * 16 + nl) * 32], h);
#pragma unroll
    for (int nt = 0; nt < 2; ++nt)
      b[nt] = load_b_frag(&lB[(wn * 32 + nt * 16 + nl) * 32 + h * 16]);
#pragma unroll
    for (int mt = 0; mt < 2; ++mt)
#pragma unroll
      for (int nt = 0; nt < 2; ++nt)
        c[mt][nt] = __builtin_amdgcn_wmma_f32_16x16x32_bf16(
            false, a[mt], false, b[nt], (short)0, c[mt][nt], false, false);
    __syncthreads();
  }

#pragma unroll
  for (int mt = 0; mt < 2; ++mt) {
#pragma unroll
    for (int nt = 0; nt < 2; ++nt) {
#pragma unroll
      for (int i = 0; i < 8; ++i) {
        int m = m0 + wm * 32 + mt * 16 + i + h * 8;
        int n = n0 + wn * 32 + nt * 16 + nl;
        float val = c[mt][nt][i] + bias[n];
        if (mode == 0) {
          int which = n >> 10;          // 0:q 1:k 2:v
          int hh = (n >> 6) & 15;
          int dd = n & 63;
          int bb = m >> 11;
          int s  = m & 2047;
          size_t bh = (size_t)(bb * H_ + hh);
          if (which == 0)      q[(bh * S_ + s) * HD_ + dd]  = (__bf16)val;
          else if (which == 1) k[(bh * S_ + s) * HD_ + dd]  = (__bf16)val;
          else                 vT[(bh * HD_ + dd) * S_ + s] = (__bf16)val;  // transposed
        } else {
          outF[(size_t)m * Ndim + n] = val;
        }
      }
    }
  }
}

// One workgroup = one (b,h) x 16 query rows.  8 waves, each owns a 256-wide
// key-column slab.  Scores -> 16x2048 f32 LDS slab, softmax, coalesced attn
// write, then PV straight from LDS (saves the 537MB attn re-read).
__global__ __launch_bounds__(256) void attn_kernel(
    const __bf16* __restrict__ qg, const __bf16* __restrict__ kg,
    const __bf16* __restrict__ vTg, float* __restrict__ attnOut,
    __bf16* __restrict__ ctxb) {
  extern __shared__ char smem[];
  float*  sc     = (float*)smem;                  // 16 x 2048 f32  (128 KB)
  __bf16* qt     = (__bf16*)(smem + 131072);      // 16 x 64 bf16   (2 KB)
  __bf16* stg    = (__bf16*)(smem + 133120);      // 8 waves x 4 KB (32 KB)
  float*  rpart  = (float*)(smem + 165888);       // 16 x 16
  float*  rowmax = (float*)(smem + 166912);       // 16
  float*  rowinv = (float*)(smem + 166976);       // 16

  const int tid   = threadIdx.x;
  const int lane  = tid & 31;
  const int h     = lane >> 4;
  const int nl    = lane & 15;
  const int wid   = tid >> 5;
  const int qbase = blockIdx.x * 16;
  const int bh    = blockIdx.y;

  const __bf16* qsrc  = qg + ((size_t)bh * S_ + qbase) * HD_;
  const __bf16* kbase = kg + (size_t)bh * S_ * HD_;
  const __bf16* vtb   = vTg + (size_t)bh * HD_ * S_;     // [64][2048]

  if (wid == 0) {       // q tile: contiguous 2KB, async DMA by wave 0
#pragma unroll
    for (int j = 0; j < 4; ++j)
      async_g2l_16B((char*)qt + j * 512 + lane * 16,
                    (const char*)qsrc + j * 512 + lane * 16);
    wait_async0();
  }
  __syncthreads();

  v16bf a0 = load_a_frag(&qt[nl * 64], h);        // k = 0..31
  v16bf a1 = load_a_frag(&qt[nl * 64 + 32], h);   // k = 32..63

  __bf16* wst = stg + wid * 2048;   // 4 KB per-wave staging

  // ---- Phase A: scores = scale * q @ k^T ----
  for (int ct = 0; ct < 16; ++ct) {
    int scol = wid * 256 + ct * 16;
    const __bf16* ks = kbase + (size_t)scol * HD_;   // 16x64 tile = 2KB contig
#pragma unroll
    for (int j = 0; j < 4; ++j)
      async_g2l_16B((char*)wst + j * 512 + lane * 16,
                    (const char*)ks + j * 512 + lane * 16);
    wait_async0();

    // B frag: n = key row (nl), k = head dim -> contiguous in [s][64] tile.
    v16bf b0 = load_b_frag(&wst[nl * 64 + h * 16]);
    v16bf b1 = load_b_frag(&wst[nl * 64 + 32 + h * 16]);
    v8f cc = {};
    cc = __builtin_amdgcn_wmma_f32_16x16x32_bf16(false, a0, false, b0,
                                                 (short)0, cc, false, false);
    cc = __builtin_amdgcn_wmma_f32_16x16x32_bf16(false, a1, false, b1,
                                                 (short)0, cc, false, false);
#pragma unroll
    for (int i = 0; i < 8; ++i)
      sc[(i + h * 8) * 2048 + scol + nl] = cc[i] * 0.125f;   // SCALE = 64^-0.5
  }
  __syncthreads();

  // ---- Softmax: 16 threads per row ----
  {
    const int r  = tid >> 4;
    const int cp = tid & 15;
    float* srow = sc + r * 2048;
    float mx = -3.4e38f;
    for (int ci = cp * 128; ci < cp * 128 + 128; ++ci) mx = fmaxf(mx, srow[ci]);
    rpart[tid] = mx;
    __syncthreads();
    if (tid < 16) {
      float m2 = -3.4e38f;
      for (int j = 0; j < 16; ++j) m2 = fmaxf(m2, rpart[tid * 16 + j]);
      rowmax[tid] = m2;
    }
    __syncthreads();
    float rm = rowmax[r];
    float sum = 0.f;
    for (int ci = cp * 128; ci < cp * 128 + 128; ++ci) {
      float e = __expf(srow[ci] - rm);
      srow[ci] = e;
      sum += e;
    }
    rpart[tid] = sum;
    __syncthreads();
    if (tid < 16) {
      float s2 = 0.f;
      for (int j = 0; j < 16; ++j) s2 += rpart[tid * 16 + j];
      rowinv[tid] = 1.0f / s2;
    }
    __syncthreads();
  }

  // ---- Write normalized attn (coalesced float4 stream; the 537MB cost) ----
  {
    float4* a4 = (float4*)(attnOut + ((size_t)bh * S_ + qbase) * S_);
    const float4* s4 = (const float4*)sc;
    for (int idx = tid; idx < 16 * 512; idx += 256) {
      float inv = rowinv[idx >> 9];
      float4 t = s4[idx];
      t.x *= inv; t.y *= inv; t.z *= inv; t.w *= inv;
      a4[idx] = t;
    }
  }

  // ---- Phase B: ctx = P @ V, k-split across the 8 waves ----
  v8f c4[4] = {};
  const float invm = rowinv[nl];
  const int kslab = wid * 256;
  for (int st = 0; st < 8; ++st) {
    int kb0 = kslab + st * 32;
    // stage vT tile [64 dd][32 s] (n-major for B frags) via async DMA
#pragma unroll
    for (int j = 0; j < 8; ++j) {
      int ii = j * 32 + lane;
      int r = ii >> 2, c8 = (ii & 3) << 3;
      async_g2l_16B((char*)wst + ii * 16, vtb + (size_t)r * S_ + kb0 + c8);
    }
    // A frag: P row (m=nl), k contiguous -> 4x float4 LDS loads + cvt
    const float* pr = sc + nl * 2048 + kb0 + h * 8;
    float4 f0 = *(const float4*)(pr);
    float4 f1 = *(const float4*)(pr + 4);
    float4 f2 = *(const float4*)(pr + 16);
    float4 f3 = *(const float4*)(pr + 20);
    v16bf a;
    a[0]  = (__bf16)(f0.x * invm); a[1]  = (__bf16)(f0.y * invm);
    a[2]  = (__bf16)(f0.z * invm); a[3]  = (__bf16)(f0.w * invm);
    a[4]  = (__bf16)(f1.x * invm); a[5]  = (__bf16)(f1.y * invm);
    a[6]  = (__bf16)(f1.z * invm); a[7]  = (__bf16)(f1.w * invm);
    a[8]  = (__bf16)(f2.x * invm); a[9]  = (__bf16)(f2.y * invm);
    a[10] = (__bf16)(f2.z * invm); a[11] = (__bf16)(f2.w * invm);
    a[12] = (__bf16)(f3.x * invm); a[13] = (__bf16)(f3.y * invm);
    a[14] = (__bf16)(f3.z * invm); a[15] = (__bf16)(f3.w * invm);
    wait_async0();
#pragma unroll
    for (int nt = 0; nt < 4; ++nt) {
      v16bf b = load_b_frag(&wst[(nt * 16 + nl) * 32 + h * 16]);
      c4[nt] = __builtin_amdgcn_wmma_f32_16x16x32_bf16(
          false, a, false, b, (short)0, c4[nt], false, false);
    }
  }
  __syncthreads();   // everyone done reading sc; reuse it for partials

  // partials stored [wid][n(64)][m(16)] so each lane stores 2x float4
#pragma unroll
  for (int nt = 0; nt < 4; ++nt) {
    float4 p0, p1;
    p0.x = c4[nt][0]; p0.y = c4[nt][1]; p0.z = c4[nt][2]; p0.w = c4[nt][3];
    p1.x = c4[nt][4]; p1.y = c4[nt][5]; p1.z = c4[nt][6]; p1.w = c4[nt][7];
    float* dstp = &sc[wid * 1024 + (nt * 16 + nl) * 16 + h * 8];
    *(float4*)(dstp)     = p0;
    *(float4*)(dstp + 4) = p1;
  }
  __syncthreads();

  const int bb = bh >> 4, hh = bh & 15;
  for (int idx = tid; idx < 1024; idx += 256) {
    float s2 = 0.f;
#pragma unroll
    for (int w2 = 0; w2 < 8; ++w2) s2 += sc[w2 * 1024 + idx];
    int n = idx >> 4, m = idx & 15;     // n = head-dim, m = q-row
    ctxb[((size_t)(bb * S_ + qbase + m)) * D_ + hh * HD_ + n] = (__bf16)s2;
  }
}

extern "C" void kernel_launch(void* const* d_in, const int* in_sizes, int n_in,
                              void* d_out, int out_size, void* d_ws,
                              size_t ws_size, hipStream_t stream) {
  (void)in_sizes; (void)n_in; (void)out_size; (void)ws_size;
  const float* x     = (const float*)d_in[0];
  const float* w_qkv = (const float*)d_in[1];
  const float* b_qkv = (const float*)d_in[2];
  const float* w_out = (const float*)d_in[3];
  const float* b_out = (const float*)d_in[4];
  float* out = (float*)d_out;

  char* ws = (char*)d_ws;                 // ~50.3 MB used
  __bf16* xb     = (__bf16*)(ws);                   // 8 MB  x bf16 [4096][1024]
  __bf16* wqkvT  = (__bf16*)(ws + 8388608);         // 6 MB  w_qkv^T [3072][1024]
  __bf16* woutT  = (__bf16*)(ws + 14680064);        // 2 MB  w_out^T [1024][1024]
  __bf16* qb     = (__bf16*)(ws + 16777216);        // 8 MB  q  [B,H,S,64]
  __bf16* kb     = (__bf16*)(ws + 25165824);        // 8 MB  k  [B,H,S,64]
  __bf16* vTb    = (__bf16*)(ws + 33554432);        // 8 MB  vT [B,H,64,S]
  __bf16* ctxb   = (__bf16*)(ws + 41943040);        // 8 MB  ctx [B,S,D]

  cvt_f32_bf16<<<1024, 256, 0, stream>>>(x, xb, M_ * D_);
  transpose_f32_bf16<<<dim3(N3_ / 32, D_ / 32), 256, 0, stream>>>(
      w_qkv, wqkvT, D_, N3_);
  transpose_f32_bf16<<<dim3(D_ / 32, D_ / 32), 256, 0, stream>>>(
      w_out, woutT, D_, D_);

  // qkv = x @ w_qkv + b_qkv -> q/k head-major, v transposed
  gemm_bf16_wmma<<<dim3(N3_ / 128, M_ / 64), 256, 0, stream>>>(
      xb, wqkvT, b_qkv, N3_, D_, /*mode=*/0, nullptr, qb, kb, vTb);

  // fused scores + softmax + attn-write + PV (163 KB dynamic LDS)
  attn_kernel<<<dim3(S_ / 16, B_ * H_), 256, 167040, stream>>>(
      qb, kb, vTb, out + (size_t)M_ * D_, ctxb);

  // out = ctx @ w_out + b_out (fp32 result)
  gemm_bf16_wmma<<<dim3(D_ / 128, M_ / 64), 256, 0, stream>>>(
      ctxb, woutT, b_out, D_, D_, /*mode=*/1, out, nullptr, nullptr, nullptr);
}
